// CRF_67138928771251
// MI455X (gfx1250) — matrix operand order repaired and checked by reading, not compiled, not run on hardware
//
#include <hip/hip_runtime.h>
#include <hip/hip_fp16.h>

// CRF NLL for MI455X (gfx1250, wave32, WMMA).
// Forward scan rewritten as max-shifted exp-domain GEMM -> v_wmma_f32_16x16x32_f16.

typedef __attribute__((ext_vector_type(16))) _Float16 v16h;
typedef __attribute__((ext_vector_type(8)))  _Float16 v8h;
typedef __attribute__((ext_vector_type(8)))  float    v8f;

#define SEQ_LEN 512
#define BSZ     1024
#define NT      64      // NUM_TAGS

// ---------------------------------------------------------------------------
// Kernel 1: numerator (gather path score), one thread per batch element.
// ---------------------------------------------------------------------------
__global__ __launch_bounds__(256)
void crf_numerator(const float* __restrict__ em,
                   const int*   __restrict__ tags,
                   const unsigned char* __restrict__ mask,
                   const float* __restrict__ start_t,
                   const float* __restrict__ end_t,
                   const float* __restrict__ trans,
                   float* __restrict__ num_out)
{
    const int b = blockIdx.x * blockDim.x + threadIdx.x;
    if (b >= BSZ) return;

    int prev = max(tags[b], 0);
    float score = start_t[prev] + em[(size_t)b * NT + prev];
    int cnt = (int)mask[b];

    for (int t = 1; t < SEQ_LEN; ++t) {
        const int   tg = max(tags[t * BSZ + b], 0);
        const float m  = (float)mask[t * BSZ + b];
        const float e  = em[((size_t)t * BSZ + b) * NT + tg];
        score += (trans[prev * NT + tg] + e) * m;
        cnt   += (int)mask[t * BSZ + b];
        prev   = tg;
    }
    const int li  = max(cnt - 1, 0);            // last_i = sum(mask) - 1
    const int ltg = max(tags[li * BSZ + b], 0);
    num_out[b] = score + end_t[ltg];
}

// ---------------------------------------------------------------------------
// Kernel 2: forward algorithm (denominator) via WMMA.
// Grid: 64 blocks (16 batch rows each), 128 threads = 4 wave32.
// Wave w owns output tag columns [16w, 16w+16); alpha lives in the WMMA
// C/D register layout: lane (l,hi), VGPR v  <->  (M = v + 8*hi, N = 16w + l).
// ---------------------------------------------------------------------------
__global__ __launch_bounds__(128)
void crf_forward_wmma(const float* __restrict__ em,
                      const unsigned char* __restrict__ mask,
                      const float* __restrict__ start_t,
                      const float* __restrict__ end_t,
                      const float* __restrict__ trans,
                      float* __restrict__ den_out)
{
    __shared__ __align__(16) _Float16 p_lds[16][NT];   // 2 KB: p matrix, A-side
    __shared__ float wred[4][16];                      // per-wave row partials

    const int tid  = threadIdx.x;
    const int wave = tid >> 5;
    const int lane = tid & 31;
    const int l    = lane & 15;
    const int hi   = lane >> 4;            // half-wave select
    const int b0   = blockIdx.x * 16;      // batch-row base of this block
    const int N    = wave * 16 + l;        // output tag column of this lane

    // ---- B fragments: exp(trans) column tile, built once, reused 511 steps.
    // 32x16 f16 B layout: lanes 0-15 hold K=0..15 (elem k -> K=k), lanes 16-31 K=16..31.
    v16h bfrag0, bfrag1;
    #pragma unroll
    for (int k = 0; k < 16; ++k) {
        bfrag0[k] = (_Float16)__expf(trans[(hi * 16 + k) * NT + N]);
        bfrag1[k] = (_Float16)__expf(trans[(32 + hi * 16 + k) * NT + N]);
    }

    // ---- alpha0 = start_transitions + emissions[0], directly in C/D layout.
    float alpha[8];
    #pragma unroll
    for (int v = 0; v < 8; ++v) {
        const int M = v + 8 * hi;
        alpha[v] = start_t[N] + em[((size_t)0 * BSZ + b0 + M) * NT + N];
    }

    // ---- sequential scan over time
    for (int t = 1; t < SEQ_LEN; ++t) {
        // (a) row max over this wave's 16 columns (butterfly within 16 lanes)
        float fm[8];
        #pragma unroll
        for (int v = 0; v < 8; ++v) {
            float x = alpha[v];
            x = fmaxf(x, __shfl_xor(x, 1, 16));
            x = fmaxf(x, __shfl_xor(x, 2, 16));
            x = fmaxf(x, __shfl_xor(x, 4, 16));
            x = fmaxf(x, __shfl_xor(x, 8, 16));
            fm[v] = x;
        }
        // (b) combine row max across the 4 waves through LDS
        if (l == 0) {
            #pragma unroll
            for (int v = 0; v < 8; ++v) wred[wave][v + 8 * hi] = fm[v];
        }
        __syncthreads();
        #pragma unroll
        for (int v = 0; v < 8; ++v) {
            const int M = v + 8 * hi;
            float x = wred[0][M];
            x = fmaxf(x, wred[1][M]);
            x = fmaxf(x, wred[2][M]);
            x = fmaxf(x, wred[3][M]);
            fm[v] = x;
        }
        // (c) p = exp(alpha - rowmax) -> LDS (f16), each wave writes its columns
        #pragma unroll
        for (int v = 0; v < 8; ++v) {
            const int M = v + 8 * hi;
            p_lds[M][N] = (_Float16)__expf(alpha[v] - fm[v]);
        }
        __syncthreads();

        // (d) A fragments (16x32 f16, this code's contiguous k order, not the ISA table order):
        //   elems 0..15 : K = hi*16 + 0..15 (matches B fragment order)
        const v8h* pr = (const v8h*)&p_lds[l][0];      // row = l; 8-half chunks
        const v8h a00 = pr[2 * hi], a01 = pr[2 * hi + 1];  // K 0..31
        const v8h a10 = pr[4 + 2 * hi], a11 = pr[5 + 2 * hi];  // K 32..63
        v16h a0, a1;
        #pragma unroll
        for (int k = 0; k < 8; ++k) {
            a0[k] = a00[k]; a0[8 + k] = a01[k];
            a1[k] = a10[k]; a1[8 + k] = a11[k];
        }

        // (e) s = p * exp(trans): two chained WMMAs, fp32 accumulate
        v8f c = {};
        c = __builtin_amdgcn_wmma_f32_16x16x32_f16(false, a0, false, bfrag0,
                                                   (short)0, c, false, false);
        c = __builtin_amdgcn_wmma_f32_16x16x32_f16(false, a1, false, bfrag1,
                                                   (short)0, c, false, false);

        // (f) alpha' = m + log(s) + em[t]  (masked per batch row)
        // 8 contiguous, 8B-aligned mask bytes for this lane's rows -> one u64 load.
        const unsigned long long mk8 =
            *(const unsigned long long*)&mask[t * BSZ + b0 + 8 * hi];
        const float* emt = em + ((size_t)t * BSZ + b0) * NT;
        #pragma unroll
        for (int v = 0; v < 8; ++v) {
            const int M = v + 8 * hi;
            const float e   = __builtin_nontemporal_load(&emt[M * NT + N]);
            const float nxt = fm[v] + __logf(c[v]) + e;
            alpha[v] = ((mk8 >> (8 * v)) & 0xffull) ? nxt : alpha[v];
        }
        // (g) prefetch next time-step's 4 KB emission tile into WGP-level cache
        if (t + 1 < SEQ_LEN)
            __builtin_prefetch(em + ((size_t)(t + 1) * BSZ + b0) * NT + tid * 8, 0, 3);
    }

    // ---- denominator = logsumexp_j(alpha + end_transitions)
    float x8[8], fm[8];
    #pragma unroll
    for (int v = 0; v < 8; ++v) x8[v] = alpha[v] + end_t[N];

    __syncthreads();                                   // wred reuse
    #pragma unroll
    for (int v = 0; v < 8; ++v) {
        float x = x8[v];
        x = fmaxf(x, __shfl_xor(x, 1, 16));
        x = fmaxf(x, __shfl_xor(x, 2, 16));
        x = fmaxf(x, __shfl_xor(x, 4, 16));
        x = fmaxf(x, __shfl_xor(x, 8, 16));
        fm[v] = x;
    }
    if (l == 0) {
        #pragma unroll
        for (int v = 0; v < 8; ++v) wred[wave][v + 8 * hi] = fm[v];
    }
    __syncthreads();
    #pragma unroll
    for (int v = 0; v < 8; ++v) {
        const int M = v + 8 * hi;
        fm[v] = fmaxf(fmaxf(wred[0][M], wred[1][M]), fmaxf(wred[2][M], wred[3][M]));
    }
    __syncthreads();                                   // all reads done before re-write
    float s8[8];
    #pragma unroll
    for (int v = 0; v < 8; ++v) {
        float x = __expf(x8[v] - fm[v]);
        x += __shfl_xor(x, 1, 16);
        x += __shfl_xor(x, 2, 16);
        x += __shfl_xor(x, 4, 16);
        x += __shfl_xor(x, 8, 16);
        s8[v] = x;
    }
    if (l == 0) {
        #pragma unroll
        for (int v = 0; v < 8; ++v) wred[wave][v + 8 * hi] = s8[v];
    }
    __syncthreads();
    if (wave == 0 && l == 0) {
        #pragma unroll
        for (int v = 0; v < 8; ++v) {
            const int M = v + 8 * hi;
            const float tot = wred[0][M] + wred[1][M] + wred[2][M] + wred[3][M];
            den_out[b0 + M] = fm[v] + __logf(tot);
        }
    }
}

// ---------------------------------------------------------------------------
// Kernel 3: nll = mean(denominator - numerator)
// ---------------------------------------------------------------------------
__global__ __launch_bounds__(256)
void crf_reduce(const float* __restrict__ num, const float* __restrict__ den,
                float* __restrict__ out)
{
    __shared__ float sm[256];
    const int tid = threadIdx.x;
    float s = 0.f;
    for (int i = tid; i < BSZ; i += 256) s += den[i] - num[i];
    sm[tid] = s;
    __syncthreads();
    for (int off = 128; off > 0; off >>= 1) {
        if (tid < off) sm[tid] += sm[tid + off];
        __syncthreads();
    }
    if (tid == 0) out[0] = sm[0] * (1.0f / (float)BSZ);
}

// ---------------------------------------------------------------------------
extern "C" void kernel_launch(void* const* d_in, const int* in_sizes, int n_in,
                              void* d_out, int out_size, void* d_ws, size_t ws_size,
                              hipStream_t stream)
{
    const float*         em      = (const float*)d_in[0];
    const int*           tags    = (const int*)d_in[1];
    const unsigned char* mask    = (const unsigned char*)d_in[2];  // jnp.bool_: 1 byte
    const float*         start_t = (const float*)d_in[3];
    const float*         end_t   = (const float*)d_in[4];
    const float*         trans   = (const float*)d_in[5];

    float* num = (float*)d_ws;        // [BSZ]
    float* den = num + BSZ;           // [BSZ]

    crf_numerator   <<<BSZ / 256, 256, 0, stream>>>(em, tags, mask, start_t, end_t, trans, num);
    crf_forward_wmma<<<BSZ / 16,  128, 0, stream>>>(em, mask, start_t, end_t, trans, den);
    crf_reduce      <<<1,         256, 0, stream>>>(num, den, (float*)d_out);
}
